// Phase60Processor_23862838297363
// MI455X (gfx1250) — matrix operand order repaired and checked by reading, not compiled, not run on hardware
//
#include <hip/hip_runtime.h>
#include <math.h>

// ---------------- problem constants ----------------
#define Bn     8
#define Sn     4096
#define Tn     77
#define INNERn 1280
#define CROSSn 2048
#define Hn     20
#define DHn    64
#define HIDn   128
#define Mfull  (Bn * Sn)   // 32768
#define MKV    (Bn * Tn)   // 616

typedef __bf16 bf16_t;
typedef __attribute__((ext_vector_type(16))) __bf16 v16bf;
typedef __attribute__((ext_vector_type(8)))  __bf16 v8bf;
typedef __attribute__((ext_vector_type(8)))  float  v8f;

// fp32 -> bf16 bits (round to nearest even), and 2x pack
__device__ __forceinline__ unsigned bfbits(float f) {
  unsigned u = __builtin_bit_cast(unsigned, f);
  return (u + 0x7FFFu + ((u >> 16) & 1u)) >> 16;
}
__device__ __forceinline__ unsigned pack2(float lo, float hi) {
  return bfbits(lo) | (bfbits(hi) << 16);
}
__device__ __forceinline__ bf16_t f2bf(float f) {
  unsigned short h = (unsigned short)bfbits(f);
  return __builtin_bit_cast(bf16_t, h);
}

__device__ __forceinline__ float lane_xor_f(float v, int msk, int lane) {
  int r = __builtin_amdgcn_ds_bpermute((lane ^ msk) << 2,
                                       __builtin_bit_cast(int, v));
  return __builtin_bit_cast(float, r);
}

__device__ __forceinline__ float gelu_exact(float x) {
  return 0.5f * x * (1.0f + erff(x * 0.70710678118654752f));
}
__device__ __forceinline__ float sigmoidf_dev(float x) {
  return 1.0f / (1.0f + __expf(-x));
}

// ---------------------------------------------------------------
// Tiled WMMA GEMM: C[M,N] (=|+=) act(A[M,K] @ W[K,N] + bias)
// Block 128 threads (4 waves), tile 128(M) x 128(N), K-step 32.
// Each wave owns a 64x64 quadrant: 4 A-frags + 4 B-frags -> 16 WMMAs
// per K-step (1 ds_load_b128 per WMMA). bf16 LDS staging with packed
// uint2/uint stores; f32 accumulate.
// GUARD=false requires M%128==0 && N%128==0 (no bounds code at all).
// flags: 1 = add bias, 2 = exact GELU on result, 4 = accumulate into C
// ---------------------------------------------------------------
#define FLAG_BIAS 1
#define FLAG_GELU 2
#define FLAG_ACC  4

template <bool GUARD>
__global__ __launch_bounds__(128)
void gemm_wmma_kernel(const float* __restrict__ A, const float* __restrict__ W,
                      const float* __restrict__ bias, float* __restrict__ C,
                      int M, int N, int K, int flags) {
  __shared__ __align__(16) bf16_t Alds[128 * 32];     // [m][k]
  __shared__ __align__(16) bf16_t Blds[128 * 32];     // transposed: [n][k]

  const int tid  = threadIdx.x;
  const int lane = tid & 31;
  const int wave = tid >> 5;
  const int n0 = blockIdx.x * 128;
  const int m0 = blockIdx.y * 128;
  const int mq = (wave >> 1) * 64;     // wave's M quadrant
  const int nq = (wave & 1) * 64;      // wave's N quadrant

  v8f acc[4][4];
  for (int i = 0; i < 4; ++i)
    for (int j = 0; j < 4; ++j)
      for (int r = 0; r < 8; ++r) acc[i][j][r] = 0.0f;

  const int kb  = (lane & 16) ? 8 : 0;    // A fragment K sub-block
  const int kb2 = (lane & 16) ? 16 : 0;   // B fragment K sub-block

  for (int kk = 0; kk < K; kk += 32) {
    __syncthreads();
    // ---- A tile 128x32 -> LDS, float4 loads, packed b64 stores ----
    for (int i = tid; i < 1024; i += 128) {
      int m = i >> 3, c = (i & 7) << 2;
      float4 v;
      if (!GUARD || (m0 + m < M)) {
        v = *(const float4*)(A + (size_t)(m0 + m) * K + kk + c);
      } else {
        v.x = v.y = v.z = v.w = 0.0f;
      }
      uint2 p; p.x = pack2(v.x, v.y); p.y = pack2(v.z, v.w);
      *(uint2*)(Alds + m * 32 + c) = p;
    }
    // ---- W tile 32x128 -> LDS transposed [n][k], pack K-pairs ----
    for (int i = tid; i < 512; i += 128) {
      int kp = (i >> 5) << 1;          // K pair base: 0,2,...,30
      int n4 = (i & 31) << 2;          // N group of 4
      float4 r0, r1;
      if (!GUARD || (n0 + n4 < N)) {
        const float* w0 = W + (size_t)(kk + kp) * N + n0 + n4;
        r0 = *(const float4*)w0;
        r1 = *(const float4*)(w0 + N);
      } else {
        r0.x = r0.y = r0.z = r0.w = 0.0f;
        r1 = r0;
      }
      *(unsigned*)(Blds + (n4 + 0) * 32 + kp) = pack2(r0.x, r1.x);
      *(unsigned*)(Blds + (n4 + 1) * 32 + kp) = pack2(r0.y, r1.y);
      *(unsigned*)(Blds + (n4 + 2) * 32 + kp) = pack2(r0.z, r1.z);
      *(unsigned*)(Blds + (n4 + 3) * 32 + kp) = pack2(r0.w, r1.w);
    }
    if (!GUARD && (kk + 32 < K))
      __builtin_prefetch(A + (size_t)(m0 + tid) * K + kk + 32, 0, 1);
    __syncthreads();

    v16bf afr[4], bfr[4];
    for (int q = 0; q < 4; ++q) {
      int row = mq + q * 16 + (lane & 15);
      ((v8bf*)&afr[q])[0] = *(const v8bf*)(Alds + row * 32 + kb);
      ((v8bf*)&afr[q])[1] = *(const v8bf*)(Alds + row * 32 + kb + 16);
      int col = nq + q * 16 + (lane & 15);
      ((v8bf*)&bfr[q])[0] = *(const v8bf*)(Blds + col * 32 + kb2 / 2);
      ((v8bf*)&bfr[q])[1] = *(const v8bf*)(Blds + col * 32 + kb2 / 2 + 16);
    }
    for (int i = 0; i < 4; ++i)
      for (int j = 0; j < 4; ++j)
        acc[i][j] = __builtin_amdgcn_wmma_f32_16x16x32_bf16(
            false, afr[i], false, bfr[j], (short)0, acc[i][j], false, false);
  }

  // writeback: VGPR r -> m = r + 8*[lane>=16]; n = lane&15
  for (int i = 0; i < 4; ++i) {
    for (int j = 0; j < 4; ++j) {
      int n = n0 + nq + j * 16 + (lane & 15);
      if (GUARD && n >= N) continue;
      for (int r = 0; r < 8; ++r) {
        int m = m0 + mq + i * 16 + r + ((lane & 16) ? 8 : 0);
        if (GUARD && m >= M) continue;
        float v = acc[i][j][r];
        if (flags & FLAG_BIAS) v += bias[n];
        if (flags & FLAG_GELU) v = gelu_exact(v);
        size_t idx = (size_t)m * N + n;
        if (flags & FLAG_ACC) C[idx] += v; else C[idx] = v;
      }
    }
  }
}

// ---------------------------------------------------------------
// Fused attention for one (b, h, 64-row s-tile):
//   scores = (Q Kt^T)*scale -> masked softmax over T -> O = P V
// NT = number of 16-wide token subtiles (QK^T N-dim = NT*16)
// KT = token dim padded to multiple of 32 for the PV K-loop
// toks == nullptr: identity token gather (global attention).
// ---------------------------------------------------------------
template <int NT, int KT>
__global__ __launch_bounds__(128)
void attend_wmma_kernel(const float* __restrict__ Q, const float* __restrict__ Kb,
                        const float* __restrict__ Vb, const int* __restrict__ toks,
                        int Tvalid, float* __restrict__ O) {
  __shared__ __align__(16) bf16_t Qlds[64 * 64];       // [s][d]
  __shared__ __align__(16) bf16_t Klds[NT * 16 * 64];  // [t][d]
  __shared__ __align__(16) bf16_t Vt[64 * KT];         // transposed: [d][t]
  __shared__ __align__(16) bf16_t Plds[64 * KT];       // probs [s][t]

  const int tid  = threadIdx.x;
  const int lane = tid & 31;
  const int wave = tid >> 5;
  const int s0 = blockIdx.x * 64;
  const int h  = blockIdx.y;
  const int b  = blockIdx.z;

  const float* Qp = Q + ((size_t)b * Sn + s0) * INNERn + h * DHn;
  const size_t kvbase = (size_t)b * Tn;

  // Q tile: float4 loads, packed b64 stores
  for (int i = tid; i < 1024; i += 128) {
    int m = i >> 4, c = (i & 15) << 2;
    float4 v = *(const float4*)(Qp + (size_t)m * INNERn + c);
    uint2 p; p.x = pack2(v.x, v.y); p.y = pack2(v.z, v.w);
    *(uint2*)(Qlds + m * 64 + c) = p;
  }
  // K tile (gathered rows), float4 loads
  for (int i = tid; i < NT * 256; i += 128) {
    int t = i >> 4, c = (i & 15) << 2;
    float4 v;
    if (t < Tvalid) {
      int tr = toks ? toks[t] : t;
      v = *(const float4*)(Kb + (kvbase + tr) * INNERn + h * DHn + c);
    } else {
      v.x = v.y = v.z = v.w = 0.0f;
    }
    uint2 p; p.x = pack2(v.x, v.y); p.y = pack2(v.z, v.w);
    *(uint2*)(Klds + t * 64 + c) = p;
  }
  // V transposed [d][t] + zero-init P (small, scalar)
  for (int i = tid; i < 64 * KT; i += 128) {
    int d = i / KT, t = i % KT;
    float v = 0.0f;
    if (t < Tvalid) {
      int tr = toks ? toks[t] : t;
      v = Vb[(kvbase + tr) * INNERn + h * DHn + d];
    }
    Vt[i] = f2bf(v);
    Plds[i] = f2bf(0.0f);   // pads columns [NT*16, KT)
  }
  __syncthreads();

  const int row = (wave << 4) + (lane & 15);
  const int kb  = (lane & 16) ? 8 : 0;
  const int kb2 = (lane & 16) ? 16 : 0;

  // ---- scores = Q (64 x 64) @ K^T (64 x NT*16) ----
  v8f sacc[NT];
  for (int t = 0; t < NT; ++t)
    for (int j = 0; j < 8; ++j) sacc[t][j] = 0.0f;
  for (int kk = 0; kk < 64; kk += 32) {
    v16bf afrag;
    ((v8bf*)&afrag)[0] = *(const v8bf*)(Qlds + row * 64 + kk + kb);
    ((v8bf*)&afrag)[1] = *(const v8bf*)(Qlds + row * 64 + kk + kb + 16);
    for (int nt = 0; nt < NT; ++nt) {
      int col = nt * 16 + (lane & 15);
      v16bf bfrag;
      ((v8bf*)&bfrag)[0] = *(const v8bf*)(Klds + col * 64 + kk + kb2 / 2);
      ((v8bf*)&bfrag)[1] = *(const v8bf*)(Klds + col * 64 + kk + kb2 / 2 + 16);
      sacc[nt] = __builtin_amdgcn_wmma_f32_16x16x32_bf16(
          false, afrag, false, bfrag, (short)0, sacc[nt], false, false);
    }
  }

  // ---- masked softmax over t (row-wise, within 16-lane halves) ----
  const float scale = 0.125f;   // 64^-0.5
  const int nbase = lane & 15;
  const int mhalf = (wave << 4) + ((lane & 16) ? 8 : 0);
  for (int r = 0; r < 8; ++r) {
    float sv[NT];
    float mx = -1e30f;
    for (int nt = 0; nt < NT; ++nt) {
      int n = nt * 16 + nbase;
      float v = (n < Tvalid) ? sacc[nt][r] * scale : -1e30f;
      sv[nt] = v;
      mx = fmaxf(mx, v);
    }
    for (int msk = 1; msk < 16; msk <<= 1) mx = fmaxf(mx, lane_xor_f(mx, msk, lane));
    float sum = 0.0f;
    for (int nt = 0; nt < NT; ++nt) {
      int n = nt * 16 + nbase;
      float e = (n < Tvalid) ? __expf(sv[nt] - mx) : 0.0f;
      sv[nt] = e;
      sum += e;
    }
    for (int msk = 1; msk < 16; msk <<= 1) sum += lane_xor_f(sum, msk, lane);
    float inv = 1.0f / sum;
    int m = mhalf + r;
    for (int nt = 0; nt < NT; ++nt)
      Plds[m * KT + nt * 16 + nbase] = f2bf(sv[nt] * inv);
  }
  __syncthreads();

  // ---- O = P (64 x KT) @ V (KT x 64) ----
  v8f oacc[4];
  for (int t = 0; t < 4; ++t)
    for (int j = 0; j < 8; ++j) oacc[t][j] = 0.0f;
  for (int kk = 0; kk < KT; kk += 32) {
    v16bf afrag;
    ((v8bf*)&afrag)[0] = *(const v8bf*)(Plds + row * KT + kk + kb);
    ((v8bf*)&afrag)[1] = *(const v8bf*)(Plds + row * KT + kk + kb + 16);
    for (int nt = 0; nt < 4; ++nt) {
      int col = nt * 16 + (lane & 15);
      v16bf bfrag;
      ((v8bf*)&bfrag)[0] = *(const v8bf*)(Vt + col * KT + kk + kb2 / 2);
      ((v8bf*)&bfrag)[1] = *(const v8bf*)(Vt + col * KT + kk + kb2 / 2 + 16);
      oacc[nt] = __builtin_amdgcn_wmma_f32_16x16x32_bf16(
          false, afrag, false, bfrag, (short)0, oacc[nt], false, false);
    }
  }

  float* Op = O + ((size_t)b * Sn + s0) * INNERn + h * DHn;
  for (int nt = 0; nt < 4; ++nt) {
    int n = nt * 16 + (lane & 15);
    for (int r = 0; r < 8; ++r) {
      int m = (wave << 4) + r + ((lane & 16) ? 8 : 0);
      Op[(size_t)m * INNERn + n] = oacc[nt][r];
    }
  }
}

// ---------------- small VALU helper kernels ----------------

__global__ void rank_project_kernel(const float* __restrict__ A,
                                    const float* __restrict__ Wa,
                                    float* __restrict__ P, int M, int K) {
  int m = blockIdx.x * blockDim.x + threadIdx.x;
  if (m >= M) return;
  float acc0 = 0.f, acc1 = 0.f, acc2 = 0.f, acc3 = 0.f;
  const float* a = A + (size_t)m * K;
  for (int k = 0; k < K; ++k) {
    float v = a[k];
    acc0 += v * Wa[k * 4 + 0]; acc1 += v * Wa[k * 4 + 1];
    acc2 += v * Wa[k * 4 + 2]; acc3 += v * Wa[k * 4 + 3];
  }
  P[(size_t)m * 4 + 0] = acc0; P[(size_t)m * 4 + 1] = acc1;
  P[(size_t)m * 4 + 2] = acc2; P[(size_t)m * 4 + 3] = acc3;
}

__global__ void rank_update_kernel(const float* __restrict__ P,
                                   const float* __restrict__ Wb,
                                   float* __restrict__ C, int M, int N) {
  size_t idx = (size_t)blockIdx.x * blockDim.x + threadIdx.x;
  if (idx >= (size_t)M * N) return;
  size_t m = idx / N; int n = (int)(idx % N);
  float s = P[m * 4 + 0] * Wb[0 * N + n] + P[m * 4 + 1] * Wb[1 * N + n] +
            P[m * 4 + 2] * Wb[2 * N + n] + P[m * 4 + 3] * Wb[3 * N + n];
  C[idx] += s;
}

__global__ void head2_kernel(const float* __restrict__ Hb,
                             const float* __restrict__ w2,
                             const float* __restrict__ b2,
                             float* __restrict__ out, int M, int do_sigmoid) {
  int m = blockIdx.x * blockDim.x + threadIdx.x;
  if (m >= M) return;
  float s = b2[0];
  const float* h = Hb + (size_t)m * HIDn;
  for (int k = 0; k < HIDn; ++k) s += h[k] * w2[k];
  out[m] = do_sigmoid ? sigmoidf_dev(s) : s;
}

__global__ void compose_kernel(const float* __restrict__ Fg,
                               const float* __restrict__ F0,
                               const float* __restrict__ F1,
                               const float* __restrict__ a0v,
                               const float* __restrict__ d0v,
                               const float* __restrict__ a1v,
                               const float* __restrict__ d1v,
                               float* __restrict__ out, int M, int N) {
  size_t idx = (size_t)blockIdx.x * blockDim.x + threadIdx.x;
  if (idx >= (size_t)M * N) return;
  size_t m = idx / N;
  float a0 = a0v[m], a1 = a1v[m], d0 = d0v[m], d1 = d1v[m];
  float front0 = sigmoidf_dev(10.0f * (d1 - d0));
  float own0 = a0 * (1.0f - a1 + a1 * front0);
  float own1 = a1 * (1.0f - a0 + a0 * (1.0f - front0));
  float bg = fminf(fmaxf(1.0f - fmaxf(a0, a1), 0.0f), 1.0f);
  float inv = 1.0f / fmaxf(own0 + own1 + bg, 1e-6f);
  out[idx] = (bg * Fg[idx] + own0 * F0[idx] + own1 * F1[idx]) * inv;
}

// ---------------- host-side orchestration ----------------
static inline void launch_gemm(const float* A, const float* W, const float* bias,
                               float* C, int M, int N, int K, int flags,
                               hipStream_t stream) {
  dim3 grid((N + 127) / 128, (M + 127) / 128);
  if ((M % 128 == 0) && (N % 128 == 0))
    gemm_wmma_kernel<false><<<grid, 128, 0, stream>>>(A, W, bias, C, M, N, K, flags);
  else
    gemm_wmma_kernel<true><<<grid, 128, 0, stream>>>(A, W, bias, C, M, N, K, flags);
}

extern "C" void kernel_launch(void* const* d_in, const int* in_sizes, int n_in,
                              void* d_out, int out_size, void* d_ws, size_t ws_size,
                              hipStream_t stream) {
  const float* hidden = (const float*)d_in[0];
  const float* enc    = (const float*)d_in[1];
  const int*   toks0  = (const int*)d_in[2];
  const int*   toks1  = (const int*)d_in[3];
  const float* Wq = (const float*)d_in[4];  const float* bq = (const float*)d_in[5];
  const float* Wk = (const float*)d_in[6];  const float* bk = (const float*)d_in[7];
  const float* Wv = (const float*)d_in[8];  const float* bv = (const float*)d_in[9];
  const float* Wo = (const float*)d_in[10]; const float* bo = (const float*)d_in[11];
  const float* lk_a = (const float*)d_in[12]; const float* lk_b = (const float*)d_in[13];
  const float* lv_a = (const float*)d_in[14]; const float* lv_b = (const float*)d_in[15];
  const float* lo_a = (const float*)d_in[16]; const float* lo_b = (const float*)d_in[17];
  const float* s0d = (const float*)d_in[18]; const float* s0u = (const float*)d_in[19];
  const float* s1d = (const float*)d_in[20]; const float* s1u = (const float*)d_in[21];
  const float* hw[16]; // e0a_w1,b1,w2,b2, e0d_..., e1a_..., e1d_...
  for (int i = 0; i < 16; ++i) hw[i] = (const float*)d_in[22 + i];

  float* out = (float*)d_out;

  // workspace carve-up (f32)
  float* ws = (float*)d_ws;
  size_t off = 0;
  float* Qb   = ws + off; off += (size_t)Mfull * INNERn;   // Q, later reused as `composed`
  float* Ktmp = ws + off; off += (size_t)MKV * INNERn;
  float* Vtmp = ws + off; off += (size_t)MKV * INNERn;
  float* Fg   = ws + off; off += (size_t)Mfull * INNERn;
  float* F0   = ws + off; off += (size_t)Mfull * INNERn;
  float* F1   = ws + off; off += (size_t)Mfull * INNERn;
  float* Hb   = ws + off; off += (size_t)Mfull * HIDn;
  float* Gb   = ws + off; off += (size_t)Mfull * 64;
  float* alpha0 = ws + off; off += Mfull;
  float* depth0 = ws + off; off += Mfull;
  float* alpha1 = ws + off; off += Mfull;
  float* depth1 = ws + off; off += Mfull;
  float* Pkv  = ws + off; off += (size_t)MKV * 4;
  float* Po   = ws + off; off += (size_t)Mfull * 4;

  // ---- K/V projection + shared LoRA ----
  launch_gemm(enc, Wk, bk, Ktmp, MKV, INNERn, CROSSn, FLAG_BIAS, stream);
  rank_project_kernel<<<(MKV + 255) / 256, 256, 0, stream>>>(enc, lk_a, Pkv, MKV, CROSSn);
  rank_update_kernel<<<((size_t)MKV * INNERn + 255) / 256, 256, 0, stream>>>(Pkv, lk_b, Ktmp, MKV, INNERn);
  launch_gemm(enc, Wv, bv, Vtmp, MKV, INNERn, CROSSn, FLAG_BIAS, stream);
  rank_project_kernel<<<(MKV + 255) / 256, 256, 0, stream>>>(enc, lv_a, Pkv, MKV, CROSSn);
  rank_update_kernel<<<((size_t)MKV * INNERn + 255) / 256, 256, 0, stream>>>(Pkv, lv_b, Vtmp, MKV, INNERn);

  // ---- Q projection ----
  launch_gemm(hidden, Wq, bq, Qb, Mfull, INNERn, INNERn, FLAG_BIAS, stream);

  // ---- attentions (global + 2 entity-masked) ----
  dim3 agrid(Sn / 64, Hn, Bn);
  attend_wmma_kernel<5, 96><<<agrid, 128, 0, stream>>>(Qb, Ktmp, Vtmp, nullptr, Tn, Fg);
  attend_wmma_kernel<1, 32><<<agrid, 128, 0, stream>>>(Qb, Ktmp, Vtmp, toks0, 8, F0);
  attend_wmma_kernel<1, 32><<<agrid, 128, 0, stream>>>(Qb, Ktmp, Vtmp, toks1, 8, F1);

  // ---- slot adapters: F += gelu(F @ down) @ up ----
  launch_gemm(F0, s0d, nullptr, Gb, Mfull, 64, INNERn, FLAG_GELU, stream);
  launch_gemm(Gb, s0u, nullptr, F0, Mfull, INNERn, 64, FLAG_ACC, stream);
  launch_gemm(F1, s1d, nullptr, Gb, Mfull, 64, INNERn, FLAG_GELU, stream);
  launch_gemm(Gb, s1u, nullptr, F1, Mfull, INNERn, 64, FLAG_ACC, stream);

  // ---- entity heads: alpha (sigmoid) and depth per entity ----
  const int hb2 = (Mfull + 255) / 256;
  launch_gemm(F0, hw[0], hw[1], Hb, Mfull, HIDn, INNERn, FLAG_BIAS | FLAG_GELU, stream);
  head2_kernel<<<hb2, 256, 0, stream>>>(Hb, hw[2], hw[3], alpha0, Mfull, 1);
  launch_gemm(F0, hw[4], hw[5], Hb, Mfull, HIDn, INNERn, FLAG_BIAS | FLAG_GELU, stream);
  head2_kernel<<<hb2, 256, 0, stream>>>(Hb, hw[6], hw[7], depth0, Mfull, 0);
  launch_gemm(F1, hw[8], hw[9], Hb, Mfull, HIDn, INNERn, FLAG_BIAS | FLAG_GELU, stream);
  head2_kernel<<<hb2, 256, 0, stream>>>(Hb, hw[10], hw[11], alpha1, Mfull, 1);
  launch_gemm(F1, hw[12], hw[13], Hb, Mfull, HIDn, INNERn, FLAG_BIAS | FLAG_GELU, stream);
  head2_kernel<<<hb2, 256, 0, stream>>>(Hb, hw[14], hw[15], depth1, Mfull, 0);

  // ---- compositing (reuse Qb as `composed`) ----
  compose_kernel<<<((size_t)Mfull * INNERn + 255) / 256, 256, 0, stream>>>(
      Fg, F0, F1, alpha0, depth0, alpha1, depth1, Qb, Mfull, INNERn);

  // ---- output projection + out-LoRA ----
  launch_gemm(Qb, Wo, bo, out, Mfull, INNERn, INNERn, FLAG_BIAS, stream);
  rank_project_kernel<<<(Mfull + 255) / 256, 256, 0, stream>>>(Qb, lo_a, Po, Mfull, INNERn);
  rank_update_kernel<<<((size_t)Mfull * INNERn + 255) / 256, 256, 0, stream>>>(Po, lo_b, out, Mfull, INNERn);
}
